// GraphChangeEncoder_1967095021706
// MI455X (gfx1250) — matrix operand order repaired and checked by reading, not compiled, not run on hardware
//
#include <hip/hip_runtime.h>
#include <math.h>

// ---- problem constants (match reference) ----
#define NN       50000
#define EE       30000
#define TT       8
#define HN       128
#define CONTENTC 96
#define TAGC     32
#define BBEX     64

// All per-wave 16-row tiles are fully in-range (wave-uniform tile checks handle tails)
static_assert(EE % 16 == 0, "edge tiles exact");
static_assert(NN % 16 == 0, "node tiles exact");

typedef __attribute__((ext_vector_type(16))) __bf16 v16bf;
typedef __attribute__((ext_vector_type(8)))  __bf16 v8bf;
typedef __attribute__((ext_vector_type(8)))  float  v8f;

// round-to-nearest-even f32 -> bf16
__device__ __forceinline__ __bf16 f2bf(float f) {
  union { float f; unsigned u; } a; a.f = f;
  unsigned r = a.u + 0x7FFFu + ((a.u >> 16) & 1u);
  unsigned short h = (unsigned short)(r >> 16);
  union { unsigned short s; __bf16 b; } c; c.s = h;
  return c.b;
}

__device__ __forceinline__ float sigm(float x) { return 1.0f / (1.0f + __expf(-x)); }

// A-operand (16x32 bf16, wave32): lane = row (lane&15); per 32-K chunk lane needs
// K = {half*8 .. half*8+7} and {16+half*8 .. 16+half*8+7}  -> two aligned 16B loads.
__device__ __forceinline__ v16bf load_a_bf(const __bf16* rowp, int kc, int half) {
  const __bf16* p = rowp + kc * 32 + half * 8;
  v8bf lo = *(const v8bf*)p;
  v8bf hi = *(const v8bf*)(p + 16);
  return __builtin_shufflevector(lo, hi, 0,1,2,3,4,5,6,7,8,9,10,11,12,13,14,15);
}

__device__ __forceinline__ v16bf load_a_f32(const float* rowp, int kc, int half) {
  const float* p = rowp + kc * 32 + half * 8;
  v16bf a;
#pragma unroll
  for (int e = 0; e < 8; ++e) { a[e] = f2bf(p[e]); a[8 + e] = f2bf(p[16 + e]); }
  return a;
}

// B-operand (32x16 bf16): lane = col (lane&15), element e -> K = half*8 + e (e<8), 16 + half*8 + (e-8) (e>=8).
// Weights are pre-packed so a lane's 16 elements are contiguous (one 32B load).
// packed idx = ((kc*Ncols + col)*2 + half)*16 + e
__device__ __forceinline__ v16bf load_b(const __bf16* Wpk, int kc, int Ncols, int col, int half) {
  return *(const v16bf*)(Wpk + (((kc * Ncols) + col) * 2 + half) * 16);
}

// ---- weight pre-pack: f32 row-major (K x Ncols) -> bf16 WMMA-B layout ----
__global__ void pack_kernel(const float* __restrict__ src, __bf16* __restrict__ dst,
                            int K, int Ncols, int total) {
  int tid = blockIdx.x * blockDim.x + threadIdx.x;
  if (tid >= total) return;
  const int msz = K * Ncols;
  const int mat = tid / msz;
  const int rem = tid % msz;
  const int e  = rem & 15;
  const int hb = (rem >> 4) & 1;
  const int r2 = rem >> 5;
  const int n  = r2 % Ncols;
  const int kc = r2 / Ncols;
  const int k  = kc * 32 + (e < 8 ? hb * 8 + e : 16 + hb * 8 + (e - 8));
  dst[tid] = f2bf(src[mat * msz + k * Ncols + n]);
}

// ---- init node embeddings: [content_table[id] || tag_embedding[tag]] ----
__global__ void init_kernel(const int* __restrict__ cid, const int* __restrict__ tgid,
                            const float* __restrict__ ctab, const float* __restrict__ temb,
                            float* __restrict__ states, __bf16* __restrict__ states_bf,
                            __bf16* __restrict__ init_bf) {
  int i = blockIdx.x * blockDim.x + threadIdx.x;
  if (i >= NN * HN) return;
  int node = i / HN, c = i % HN;
  float v = (c < CONTENTC) ? ctab[cid[node] * CONTENTC + c]
                           : temb[tgid[node] * TAGC + (c - CONTENTC)];
  states[i] = v;
  __bf16 b = f2bf(v);
  states_bf[i] = b;
  init_bf[i]   = b;
}

__global__ void zero4_kernel(float4* __restrict__ p, int n4) {
  int i = blockIdx.x * blockDim.x + threadIdx.x;
  if (i < n4) p[i] = make_float4(0.f, 0.f, 0.f, 0.f);
}

__global__ void copy4_kernel(const float4* __restrict__ s, float4* __restrict__ d, int n4) {
  int i = blockIdx.x * blockDim.x + threadIdx.x;
  if (i < n4) d[i] = s[i];
}

// ---- edge messages: gather src states, per-type 128x128 GEMM via WMMA, atomic scatter ----
// grid: (ceil(E/128), T), block 256 (8 waves); each wave: 16 edges x 128 out cols
__global__ void __launch_bounds__(256) msg_kernel(
    const __bf16* __restrict__ states_bf,
    const int*    __restrict__ edge_src,
    const int*    __restrict__ edge_tgt,
    const __bf16* __restrict__ Wpk,   // [T][H*H] packed (this layer)
    const float*  __restrict__ bvec,  // [T][H]           (this layer)
    float*        __restrict__ agg) {
  const int t    = blockIdx.y;
  const int lane = threadIdx.x & 31;
  const int tile = blockIdx.x * 8 + (threadIdx.x >> 5);
  const int e0   = tile * 16;
  if (e0 >= EE) return;                 // uniform per wave -> WMMA EXEC stays all-1s
  const int half = lane >> 4;
  const int m    = lane & 15;

  // EE % 16 == 0 and e0 < EE  ->  all 16 edges of this tile are in range
  const int src = edge_src[t * EE + (e0 + m)];
  const __bf16* srow = states_bf + src * HN;

  v16bf A[4];
#pragma unroll
  for (int kc = 0; kc < 4; ++kc) A[kc] = load_a_bf(srow, kc, half);

  // 32-bit element offsets into agg (max N*H = 6.4M < 2^31)
  int tgoff[8];
#pragma unroll
  for (int i = 0; i < 8; ++i)
    tgoff[i] = edge_tgt[t * EE + (e0 + half * 8 + i)] * HN;

  const __bf16* Wt = Wpk + t * (HN * HN);
#pragma unroll
  for (int nt = 0; nt < 8; ++nt) {
    v8f acc = {0.f,0.f,0.f,0.f,0.f,0.f,0.f,0.f};
#pragma unroll
    for (int kc = 0; kc < 4; ++kc) {
      v16bf Bv = load_b(Wt, kc, HN, nt * 16 + m, half);
      acc = __builtin_amdgcn_wmma_f32_16x16x32_bf16(false, A[kc], false, Bv,
                                                    (short)0, acc, false, false);
    }
    const int   col  = nt * 16 + m;
    const float bias = bvec[t * HN + col];
#pragma unroll
    for (int i = 0; i < 8; ++i)   // D: VGPR i -> tile row (half*8 + i), col = lane&15
      atomicAdd(agg + (tgoff[i] + col), acc[i] + bias);
  }
}

// ---- fused GRU: per-wave 16-row tile; WMMA for x@Wih and h@Whh, gates in-register ----
template <int LAYER>
__global__ void __launch_bounds__(256) gru_kernel(
    const float*  __restrict__ agg,
    const __bf16* __restrict__ init_bf,     // used only for LAYER==1
    __bf16*       __restrict__ states_bf,   // read (A_h) then updated in place
    float*        __restrict__ states,      // read (h) then updated in place
    const __bf16* __restrict__ Wih_pk,      // [(LAYER?256:128) x 384] packed
    const __bf16* __restrict__ Whh_pk,      // [128 x 384] packed
    const float*  __restrict__ bih,
    const float*  __restrict__ bhh) {
  constexpr int KCI = LAYER ? 8 : 4;
  const int lane = threadIdx.x & 31;
  const int tile = blockIdx.x * 8 + (threadIdx.x >> 5);
  const int r0   = tile * 16;
  if (r0 >= NN) return;                 // uniform per wave
  const int half = lane >> 4;
  const int m    = lane & 15;
  const int row  = r0 + m;              // NN % 16 == 0 -> always in range

  v16bf Ah[4];
#pragma unroll
  for (int kc = 0; kc < 4; ++kc)
    Ah[kc] = load_a_bf(states_bf + row * HN, kc, half);

  v16bf Ai[KCI];
  if (LAYER == 0) {
#pragma unroll
    for (int kc = 0; kc < 4; ++kc)
      Ai[kc] = load_a_f32(agg + row * HN, kc, half);
  } else {
#pragma unroll
    for (int kc = 0; kc < 4; ++kc)
      Ai[kc] = load_a_bf(init_bf + row * HN, kc, half);
#pragma unroll
    for (int kc = 4; kc < KCI; ++kc)
      Ai[kc] = load_a_f32(agg + row * HN, kc - 4, half);
  }

  const int rowout = (r0 + half * 8) * HN;   // 32-bit base offset of this lane's 8 rows

#pragma unroll
  for (int j = 0; j < 8; ++j) {           // 16-col block within H
    v8f gi[3], gh[3];
#pragma unroll
    for (int g = 0; g < 3; ++g) {
      gi[g] = (v8f){0.f,0.f,0.f,0.f,0.f,0.f,0.f,0.f};
      gh[g] = (v8f){0.f,0.f,0.f,0.f,0.f,0.f,0.f,0.f};
    }
#pragma unroll
    for (int kc = 0; kc < KCI; ++kc)
#pragma unroll
      for (int g = 0; g < 3; ++g) {       // gate col base: r=0, z=128, n=256
        v16bf Bv = load_b(Wih_pk, kc, 384, g * 128 + j * 16 + m, half);
        gi[g] = __builtin_amdgcn_wmma_f32_16x16x32_bf16(false, Ai[kc], false, Bv,
                                                        (short)0, gi[g], false, false);
      }
#pragma unroll
    for (int kc = 0; kc < 4; ++kc)
#pragma unroll
      for (int g = 0; g < 3; ++g) {
        v16bf Bv = load_b(Whh_pk, kc, 384, g * 128 + j * 16 + m, half);
        gh[g] = __builtin_amdgcn_wmma_f32_16x16x32_bf16(false, Ah[kc], false, Bv,
                                                        (short)0, gh[g], false, false);
      }
    const int   col = j * 16 + m;
    const float bir = bih[col], biz = bih[128 + col], bin = bih[256 + col];
    const float bhr = bhh[col], bhz = bhh[128 + col], bhn = bhh[256 + col];
#pragma unroll
    for (int i = 0; i < 8; ++i) {
      float r  = sigm(gi[0][i] + bir + gh[0][i] + bhr);
      float z  = sigm(gi[1][i] + biz + gh[1][i] + bhz);
      float nn = tanhf(gi[2][i] + bin + r * (gh[2][i] + bhn));
      const int idx = rowout + i * HN + col;   // 32-bit element offset
      float h   = states[idx];
      float out = (1.f - z) * nn + z * h;
      states[idx]    = out;     // rows are wave-private -> race-free in-place update
      states_bf[idx] = f2bf(out);
    }
  }
}

// ---- gated pooled readout: one wave per node ----
__global__ void __launch_bounds__(256) readout_kernel(
    const float* __restrict__ s, const float* __restrict__ w, const float* __restrict__ bptr,
    const int* __restrict__ n2e, float* __restrict__ out, int coloff) {
  const int lane = threadIdx.x & 31;
  const int node = blockIdx.x * 8 + (threadIdx.x >> 5);
  if (node >= NN) return;
  const float* srow = s + node * HN;
  float dot = 0.f;
#pragma unroll
  for (int c = lane; c < HN; c += 32) dot += srow[c] * w[c];
#pragma unroll
  for (int off = 16; off > 0; off >>= 1) dot += __shfl_xor(dot, off, 32);
  const float g = sigm(dot + bptr[0]);
  const int bo = n2e[node] * (2 * HN) + coloff;
#pragma unroll
  for (int c = lane; c < HN; c += 32)
    atomicAdd(out + (bo + c), g * srow[c]);
}

extern "C" void kernel_launch(void* const* d_in, const int* in_sizes, int n_in,
                              void* d_out, int out_size, void* d_ws, size_t ws_size,
                              hipStream_t stream) {
  (void)in_sizes; (void)n_in; (void)out_size; (void)ws_size;
  const int*   cid   = (const int*)  d_in[0];
  const int*   tgid  = (const int*)  d_in[1];
  const int*   esrc  = (const int*)  d_in[2];
  const int*   etgt  = (const int*)  d_in[3];
  const int*   n2e   = (const int*)  d_in[4];
  const float* ctab  = (const float*)d_in[5];
  const float* temb  = (const float*)d_in[6];
  const float* msgW  = (const float*)d_in[7];
  const float* msgb  = (const float*)d_in[8];
  const float* g0ih  = (const float*)d_in[9];
  const float* g0hh  = (const float*)d_in[10];
  const float* g0bih = (const float*)d_in[11];
  const float* g0bhh = (const float*)d_in[12];
  const float* g1ih  = (const float*)d_in[13];
  const float* g1hh  = (const float*)d_in[14];
  const float* g1bih = (const float*)d_in[15];
  const float* g1bhh = (const float*)d_in[16];
  const float* r1w   = (const float*)d_in[17];
  const float* r1b   = (const float*)d_in[18];
  const float* r2w   = (const float*)d_in[19];
  const float* r2b   = (const float*)d_in[20];
  float* out = (float*)d_out;

  // ---- workspace carve-out (~104 MB) ----
  size_t off = 0;
  auto alloc = [&](size_t bytes) -> void* {
    off = (off + 255) & ~(size_t)255;
    void* p = (char*)d_ws + off;
    off += bytes;
    return p;
  };
  float*  states    = (float*) alloc((size_t)NN * HN * 4);
  float*  s1        = (float*) alloc((size_t)NN * HN * 4);
  float*  agg       = (float*) alloc((size_t)NN * HN * 4);
  __bf16* states_bf = (__bf16*)alloc((size_t)NN * HN * 2);
  __bf16* init_bf   = (__bf16*)alloc((size_t)NN * HN * 2);
  __bf16* msgW_pk   = (__bf16*)alloc((size_t)2 * TT * HN * HN * 2);
  __bf16* g0Wih_pk  = (__bf16*)alloc((size_t)HN * 384 * 2);
  __bf16* g0Whh_pk  = (__bf16*)alloc((size_t)HN * 384 * 2);
  __bf16* g1Wih_pk  = (__bf16*)alloc((size_t)2 * HN * 384 * 2);
  __bf16* g1Whh_pk  = (__bf16*)alloc((size_t)HN * 384 * 2);

  // ---- pack weights to bf16 WMMA-B layout ----
  {
    int tot = 2 * TT * HN * HN;   // 16 matrices of 128x128
    pack_kernel<<<(tot + 255) / 256, 256, 0, stream>>>(msgW, msgW_pk, HN, HN, tot);
    tot = HN * 384;
    pack_kernel<<<(tot + 255) / 256, 256, 0, stream>>>(g0ih, g0Wih_pk, HN, 384, tot);
    pack_kernel<<<(tot + 255) / 256, 256, 0, stream>>>(g0hh, g0Whh_pk, HN, 384, tot);
    pack_kernel<<<(tot + 255) / 256, 256, 0, stream>>>(g1hh, g1Whh_pk, HN, 384, tot);
    tot = 2 * HN * 384;
    pack_kernel<<<(tot + 255) / 256, 256, 0, stream>>>(g1ih, g1Wih_pk, 2 * HN, 384, tot);
  }

  // ---- initial node embeddings ----
  init_kernel<<<(NN * HN + 255) / 256, 256, 0, stream>>>(cid, tgid, ctab, temb,
                                                         states, states_bf, init_bf);

  const int msgBlocks = (EE / 16 + 7) / 8;   // 235
  const int gruBlocks = (NN / 16 + 7) / 8;   // 391
  const int aggVec4   = NN * HN / 4;         // 1,600,000

  for (int layer = 0; layer < 2; ++layer) {
    const __bf16* Wl = msgW_pk + (size_t)layer * TT * HN * HN;
    const float*  bl = msgb    + (size_t)layer * TT * HN;
    for (int s = 0; s < 3; ++s) {
      zero4_kernel<<<aggVec4 / 256, 256, 0, stream>>>((float4*)agg, aggVec4);
      msg_kernel<<<dim3(msgBlocks, TT), 256, 0, stream>>>(states_bf, esrc, etgt, Wl, bl, agg);
      if (layer == 0)
        gru_kernel<0><<<gruBlocks, 256, 0, stream>>>(agg, (const __bf16*)nullptr,
            states_bf, states, g0Wih_pk, g0Whh_pk, g0bih, g0bhh);
      else
        gru_kernel<1><<<gruBlocks, 256, 0, stream>>>(agg, init_bf,
            states_bf, states, g1Wih_pk, g1Whh_pk, g1bih, g1bhh);
    }
    if (layer == 0)
      copy4_kernel<<<aggVec4 / 256, 256, 0, stream>>>((const float4*)states, (float4*)s1, aggVec4);
  }

  // ---- readout ----
  {
    int n4 = BBEX * 2 * HN / 4;   // 4096
    zero4_kernel<<<(n4 + 255) / 256, 256, 0, stream>>>((float4*)out, n4);
  }
  readout_kernel<<<(NN + 7) / 8, 256, 0, stream>>>(s1,     r1w, r1b, n2e, out, 0);
  readout_kernel<<<(NN + 7) / 8, 256, 0, stream>>>(states, r2w, r2b, n2e, out, HN);
}
